// SpatialGNN_45432164057449
// MI455X (gfx1250) — matrix operand order repaired and checked by reading, not compiled, not run on hardware
//
#include <hip/hip_runtime.h>

// Problem constants (from reference): B=16, C=64, N=32, K=16, L=24, E=160, H=4, NLAYERS=2
#define C_   64
#define N_   32
#define E_   160
#define H_   4
#define D_   16
#define NB_  16
#define NK_  16
#define NLd_ 24
#define G_   (NB_ * NK_ * NLd_)   // 6144 graphs
#define NLAYERS_ 2

// LDS strides (elements). bf16 strides multiple of 8 (16B alignment for b128 loads).
#define HSTR 72   // h matrix, bf16, [32][HSTR]
#define WSTR 72   // weight matrices, bf16, [64][WSTR], row-major W[c][k]
#define XSTR 68   // f32 matrices xl/xr/agg, [32][XSTR]

typedef __attribute__((ext_vector_type(16))) __bf16         v16bf;
typedef __attribute__((ext_vector_type(8)))  float          v8f;
typedef __attribute__((ext_vector_type(8)))  unsigned short v8u16;
typedef __attribute__((ext_vector_type(16))) unsigned short v16u16;

static __device__ __forceinline__ unsigned short f2bf(float f) {
  // round-to-nearest-even fp32 -> bf16
  unsigned int u = __float_as_uint(f);
  u += 0x7fffu + ((u >> 16) & 1u);
  return (unsigned short)(u >> 16);
}

// Load a 16-element bf16 fragment as two 16-byte LDS loads.
// Elements 0..7 at p[0..7], elements 8..15 at p[c2+0..7].
static __device__ __forceinline__ v16bf load_frag(const unsigned short* p, int c2) {
  v8u16 a = *(const v8u16*)(p);
  v8u16 b = *(const v8u16*)(p + c2);
  v16u16 u;
#pragma unroll
  for (int j = 0; j < 8; ++j) { u[j] = a[j]; u[8 + j] = b[j]; }
  return __builtin_bit_cast(v16bf, u);
}

// One 16x16 output tile of  X = h(32x64,bf16) @ W^T(64x64) + bias,  W stored row-major W[c][k].
// A-fragment (16x32 bf16, ISA 7.12.2): lane<16 -> M=lane, K = kb + {0..7, 16..23}
//                                      lane>=16 -> M=lane-16, K = kb + {8..15, 24..31}
// B-fragment (32x16 bf16): lane<16 -> N=lane, K = kb + {0..7, 16..23} ; lane>=16 -> K = kb + {8..15, 24..31}
static __device__ __forceinline__ void gemm_tile(const unsigned short* __restrict__ hA,
                                                 const unsigned short* __restrict__ W,
                                                 const float* __restrict__ bias,
                                                 float* __restrict__ Xout,
                                                 int mBase, int nBase, int lane) {
  const int hi   = lane >> 4;      // 0 or 1
  const int lrow = lane & 15;
  v8f acc = {};
#pragma unroll
  for (int kb = 0; kb < C_; kb += 32) {
    v16bf a = load_frag(hA + (mBase + lrow) * HSTR + kb + hi * 8, 16);
    v16bf b = load_frag(W   + (nBase + lrow) * WSTR + kb + hi * 8, 16);
    acc = __builtin_amdgcn_wmma_f32_16x16x32_bf16(
        /*neg_a=*/false, a, /*neg_b=*/false, b,
        /*c_mod=*/(short)0, acc, /*reuse_a=*/false, /*reuse_b=*/false);
  }
  const float bv = bias[nBase + lrow];
#pragma unroll
  for (int r = 0; r < 8; ++r) {
    // D layout: VGPR r -> row M = mBase + r + 8*hi, col N = nBase + lrow
    Xout[(mBase + r + hi * 8) * XSTR + nBase + lrow] = acc[r] + bv;
  }
}

__global__ __launch_bounds__(128)
void SpatialGNN_gatv2_fused_kernel(const float* __restrict__ x,
                                   const int*   __restrict__ edge_index,
                                   const float* __restrict__ edge_weight,
                                   const float* __restrict__ Wl,
                                   const float* __restrict__ bl,
                                   const float* __restrict__ Wr,
                                   const float* __restrict__ br,
                                   const float* __restrict__ We,
                                   const float* __restrict__ att,
                                   const float* __restrict__ bias_out,
                                   float* __restrict__ out) {
  __shared__ __align__(16) unsigned short sh_h [N_ * HSTR];  // bf16 activations
  __shared__ __align__(16) unsigned short sh_wl[C_ * WSTR];  // bf16 Wl (row-major [c][k])
  __shared__ __align__(16) unsigned short sh_wr[C_ * WSTR];  // bf16 Wr
  __shared__ float sh_xl [N_ * XSTR];
  __shared__ float sh_xr [N_ * XSTR];
  __shared__ float sh_agg[N_ * XSTR];
  __shared__ float sh_sc [E_ * H_];        // score, then alpha
  __shared__ float sh_smax[N_ * H_];
  __shared__ float sh_den [N_ * H_];
  __shared__ int   sh_src[E_];
  __shared__ int   sh_dst[E_];
  __shared__ float sh_ew [E_];
  __shared__ float sWe[C_], sAtt[C_], sBl[C_], sBr[C_], sBias[C_];

  const int tid  = threadIdx.x;
  const int wave = tid >> 5;
  const int lane = tid & 31;
  const int g    = blockIdx.x;
  const int b    = g / (NK_ * NLd_);
  const int rem  = g - b * (NK_ * NLd_);
  const int kk   = rem / NLd_;
  const int ll   = rem - kk * NLd_;

  // ---- load h[g] = x[b, :, :, kk, ll] (gather) + zero agg ----
  for (int idx = tid; idx < N_ * C_; idx += 128) {
    const int n = idx >> 6, c = idx & 63;
    const float v = x[(((b * C_ + c) * N_ + n) * NK_ + kk) * NLd_ + ll];
    sh_h[n * HSTR + c]  = f2bf(v);
    sh_agg[n * XSTR + c] = 0.0f;
  }
  for (int e = tid; e < E_; e += 128) {
    sh_src[e] = edge_index[e];
    sh_dst[e] = edge_index[E_ + e];
    sh_ew[e]  = edge_weight[e];
  }
  // warm L2 with layer-1 weights while layer-0 runs
  __builtin_prefetch(Wl + C_ * C_, 0, 0);
  __builtin_prefetch(Wr + C_ * C_, 0, 0);

#pragma unroll
  for (int layer = 0; layer < NLAYERS_; ++layer) {
    const float* Wlp = Wl + layer * C_ * C_;
    const float* Wrp = Wr + layer * C_ * C_;
    for (int idx = tid; idx < C_ * C_; idx += 128) {
      const int c = idx >> 6, k = idx & 63;      // coalesced in k
      sh_wl[c * WSTR + k] = f2bf(Wlp[idx]);
      sh_wr[c * WSTR + k] = f2bf(Wrp[idx]);
    }
    if (tid < C_) {
      sBl[tid]   = bl[layer * C_ + tid];
      sBr[tid]   = br[layer * C_ + tid];
      sBias[tid] = bias_out[layer * C_ + tid];
      sWe[tid]   = We[layer * C_ + tid];     // We[i][:,0]
      sAtt[tid]  = att[layer * C_ + tid];    // [H][D] flattened
    }
    __syncthreads();

    // ---- GEMMs: xl = h@Wl^T + bl, xr = h@Wr^T + br.  16 tiles over 4 waves ----
#pragma unroll
    for (int t4 = 0; t4 < 4; ++t4) {
      const int t     = wave * 4 + t4;           // wave-uniform
      const int mBase = ((t >> 2) & 1) * 16;
      const int nBase = (t & 3) * 16;
      if (t < 8) gemm_tile(sh_h, sh_wl, sBl, sh_xl, mBase, nBase, lane);
      else       gemm_tile(sh_h, sh_wr, sBr, sh_xr, mBase, nBase, lane);
    }
    __syncthreads();

    // ---- attention scores: per (e, head) ----
    for (int idx = tid; idx < E_ * H_; idx += 128) {
      const int e = idx >> 2, hh = idx & 3;
      const int s = sh_src[e], d = sh_dst[e];
      const float ew = sh_ew[e];
      float acc = 0.0f;
#pragma unroll
      for (int dd = 0; dd < D_; ++dd) {
        const int c = hh * D_ + dd;
        float z = sh_xl[s * XSTR + c] + sh_xr[d * XSTR + c] + ew * sWe[c];
        z = (z > 0.0f) ? z : 0.2f * z;           // leaky_relu(0.2)
        acc += z * sAtt[c];
      }
      sh_sc[idx] = acc;
    }
    __syncthreads();

    // ---- segment max + denom per (node, head): exactly 128 items ----
    {
      const int n = tid >> 2, hh = tid & 3;
      float m = -1e30f;
      for (int e = 0; e < E_; ++e)
        if (sh_dst[e] == n) m = fmaxf(m, sh_sc[e * H_ + hh]);
      float den = 0.0f;
      for (int e = 0; e < E_; ++e)
        if (sh_dst[e] == n) den += __expf(sh_sc[e * H_ + hh] - m);
      sh_smax[tid] = m;
      sh_den[tid]  = (den > 0.0f) ? den : 1.0f;
    }
    __syncthreads();

    // ---- alpha (overwrite scores) ----
    for (int idx = tid; idx < E_ * H_; idx += 128) {
      const int e = idx >> 2, hh = idx & 3;
      const int d = sh_dst[e];
      sh_sc[idx] = __expf(sh_sc[idx] - sh_smax[d * H_ + hh]) / sh_den[d * H_ + hh];
    }
    __syncthreads();

    // ---- aggregate: agg[dst] += alpha * xl[src]  (ds_add_f32 scatter) ----
    for (int idx = tid; idx < E_ * C_; idx += 128) {
      const int e = idx >> 6, c = idx & 63, hh = c >> 4;
      const int s = sh_src[e], d = sh_dst[e];
      atomicAdd(&sh_agg[d * XSTR + c], sh_sc[e * H_ + hh] * sh_xl[s * XSTR + c]);
    }
    __syncthreads();

    // ---- finalize: +bias, (relu + requantize) or write out ----
    for (int idx = tid; idx < N_ * C_; idx += 128) {
      const int n = idx >> 6, c = idx & 63;
      float v = sh_agg[n * XSTR + c] + sBias[c];
      if (layer < NLAYERS_ - 1) {
        v = fmaxf(v, 0.0f);
        sh_h[n * HSTR + c]   = f2bf(v);
        sh_agg[n * XSTR + c] = 0.0f;            // re-zero for next layer
      } else {
        out[(((b * C_ + c) * N_ + n) * NK_ + kk) * NLd_ + ll] = v;
      }
    }
    __syncthreads();
  }
}

extern "C" void kernel_launch(void* const* d_in, const int* in_sizes, int n_in,
                              void* d_out, int out_size, void* d_ws, size_t ws_size,
                              hipStream_t stream) {
  (void)in_sizes; (void)n_in; (void)out_size; (void)d_ws; (void)ws_size;
  const float* x    = (const float*)d_in[0];
  const int*   ei   = (const int*)  d_in[1];
  const float* ew   = (const float*)d_in[2];
  const float* Wl   = (const float*)d_in[3];
  const float* bl   = (const float*)d_in[4];
  const float* Wr   = (const float*)d_in[5];
  const float* br   = (const float*)d_in[6];
  const float* We   = (const float*)d_in[7];
  const float* att  = (const float*)d_in[8];
  const float* bias = (const float*)d_in[9];
  float* out = (float*)d_out;

  hipLaunchKernelGGL(SpatialGNN_gatv2_fused_kernel,
                     dim3(G_), dim3(128), 0, stream,
                     x, ei, ew, Wl, bl, Wr, br, We, att, bias, out);
}
